// QNet_68667937128869
// MI455X (gfx1250) — matrix operand order repaired and checked by reading, not compiled, not run on hardware
//
#include <hip/hip_runtime.h>
#include <hip/hip_bf16.h>

// ---------------------------------------------------------------------------
// QNet forward for MI455X (gfx1250, wave32, WMMA).
// All K=128 GEMMs run on v_wmma_f32_16x16x32_bf16 from bf16 LDS tiles.
// A-side tiles are row-major [row][k]  -> fragment = 2x ds_load_b128 / lane.
// B-side tiles are transposed [col][k] -> fragment = 2x ds_load_b128 / lane.
// SGFormer global linear attention is fused (Q/K/V never hit HBM).
// Workspace requirement: ~170 MB (5 * BN*D f32 buffers + ~1 MB stats).
// ---------------------------------------------------------------------------

#define DEV __device__ __forceinline__

typedef __attribute__((ext_vector_type(16))) __bf16          v16bf;
typedef __attribute__((ext_vector_type(16))) unsigned short  v16u;
typedef __attribute__((ext_vector_type(8)))  unsigned short  v8u;
typedef __attribute__((ext_vector_type(4)))  unsigned short  v4u;
typedef __attribute__((ext_vector_type(8)))  float           v8f;

constexpr int Bc = 64, Nnodes = 1024, BNc = 65536, Dc = 128, Kdeg = 16;
constexpr int Hc = 4, VPc = 128, FFc = 512;
constexpr int ST     = 136;        // LDS row stride in ushorts (272B: 16B aligned, bank-skewed)
constexpr int TILE_U = 128 * ST;   // one 128x128 bf16 tile in LDS (ushort elements)

DEV unsigned short f2bf(float f){
  union { float f; unsigned u; } x; x.f = f;
  unsigned r = x.u + 0x7fffu + ((x.u >> 16) & 1u);   // RNE
  return (unsigned short)(r >> 16);
}
DEV float bf2f(unsigned short h){
  union { unsigned u; float f; } x; x.u = ((unsigned)h) << 16; return x.f;
}

DEV v8f wmma_bf16(v16bf a, v16bf b, v8f c){
  // 8 args: (neg_a, A, neg_b, B, c_mod, C, reuse_a, reuse_b)
  return __builtin_amdgcn_wmma_f32_16x16x32_bf16(false, a, false, b, (short)0, c, false, false);
}

// A fragment (16x32 bf16) from a row-major [row][k] LDS tile.
// lane m = lane&15; K runs: [kk+kh .. +7] and [kk+16+kh .. +7], kh = 8*(lane>=16).
DEV v16bf load_a(const unsigned short* s, int mbase, int kk, int lane){
  const int m  = mbase + (lane & 15);
  const int kh = (lane >> 4) * 8;
  const unsigned short* p = s + m * ST + kk + kh;
  v8u lo = *(const v8u*)(p);
  v8u hi = *(const v8u*)(p + 16);
  v16u u = __builtin_shufflevector(lo, hi, 0,1,2,3,4,5,6,7,8,9,10,11,12,13,14,15);
  return __builtin_bit_cast(v16bf, u);
}
// B fragment (32x16 bf16) from a TRANSPOSED [col][k] LDS tile.
// lane n = nbase + (lane&15); K runs: [kk+kh .. +7] and [kk+16+kh .. +7], kh = 8*(lane>=16).
DEV v16bf load_bT(const unsigned short* s, int kk, int nbase, int lane){
  const int n  = nbase + (lane & 15);
  const int kh = (lane >> 4) * 8;
  const unsigned short* p = s + n * ST + kk + kh;
  v8u lo = *(const v8u*)(p);
  v8u hi = *(const v8u*)(p + 16);
  v16u u = __builtin_shufflevector(lo, hi, 0,1,2,3,4,5,6,7,8,9,10,11,12,13,14,15);
  return __builtin_bit_cast(v16bf, u);
}

// 128x128 (MxN) += A[128xK=128] @ B ; A row-major tile, B transposed tile.
DEV void mm_tile(const unsigned short* sA, const unsigned short* sBT,
                 int lane, int mbase, v8f acc[8]){
#pragma unroll
  for (int kk = 0; kk < 128; kk += 32){
    v16bf a = load_a(sA, mbase, kk, lane);
#pragma unroll
    for (int nt = 0; nt < 8; ++nt){
      v16bf b = load_bT(sBT, kk, nt * 16, lane);
      acc[nt] = wmma_bf16(a, b, acc[nt]);
    }
  }
}

// Stage 128x128 f32 (row stride ld) -> row-major bf16 LDS tile.
// Coalesced b128 global loads, packed ds_store_b64.
DEV void stage_rowmajor(unsigned short* dst, const float* src, int ld, int tid){
  for (int e = tid; e < 128 * 32; e += 256){
    const int r = e >> 5, c4 = (e & 31) << 2;
    const float4 v = *(const float4*)(src + (long)r * ld + c4);
    v4u p; p[0] = f2bf(v.x); p[1] = f2bf(v.y); p[2] = f2bf(v.z); p[3] = f2bf(v.w);
    *(v4u*)(dst + r * ST + c4) = p;
  }
}
// Stage 128x128 f32 (row stride ld) -> TRANSPOSED [col][k] bf16 LDS tile.
// Consecutive lanes read consecutive columns (coalesced b32), packed b64 stores.
DEV void stage_colmajor(unsigned short* dst, const float* src, int ld, int tid){
  for (int e = tid; e < 128 * 32; e += 256){
    const int n = e & 127, k4 = (e >> 7) << 2;
    v4u p;
#pragma unroll
    for (int j = 0; j < 4; ++j) p[j] = f2bf(src[(long)(k4 + j) * ld + n]);
    *(v4u*)(dst + n * ST + k4) = p;
  }
}
// Stage a dense 128x128 ushort tile into padded LDS rows (b128 copy).
DEV void stage_u16(unsigned short* dst, const unsigned short* src, int tid){
  for (int e = tid; e < 128 * 16; e += 256){
    const int r = e >> 4, c8 = (e & 15) << 3;
    *(v8u*)(dst + r * ST + c8) = *(const v8u*)(src + r * 128 + c8);
  }
}
// Store a WMMA C tile (rows=M over r, cols via lane) TRANSPOSED into [col][row]
// LDS tile: per lane the 8 r-values are contiguous -> one ds_store_b128 per nt.
DEV void store_cT(unsigned short* dst, const v8f acc[8], const float* bias,
                  int wv, int lane){
  const int cb = lane & 15, rofs = (lane >> 4) * 8;
#pragma unroll
  for (int nt = 0; nt < 8; ++nt){
    const float bia = bias ? bias[nt * 16 + cb] : 0.f;
    v8u p;
#pragma unroll
    for (int r = 0; r < 8; ++r) p[r] = f2bf(acc[nt][r] + bia);
    *(v8u*)(dst + (nt * 16 + cb) * ST + wv * 16 + rofs) = p;
  }
}

// ---------------------------------------------------------------------------
// Generic C[M x 128] = act(A[M x 128] @ W[128 x 128] + bias). grid.x = M/128.
// ---------------------------------------------------------------------------
__global__ __launch_bounds__(256) void gemm128_bf16(const float* __restrict__ A,
                                                    const float* __restrict__ W,
                                                    const float* __restrict__ bias,
                                                    float* __restrict__ C, int relu){
  extern __shared__ unsigned short sm[];
  unsigned short* sA  = sm;
  unsigned short* sWT = sm + TILE_U;
  const int tid = threadIdx.x, lane = tid & 31, wv = tid >> 5;
  const int cb = lane & 15, rofs = (lane >> 4) * 8;
  const long m0 = (long)blockIdx.x * 128;

  stage_rowmajor(sA, A + m0 * 128, 128, tid);
  stage_colmajor(sWT, W, 128, tid);
  __syncthreads();

  const v8f vz = {0.f,0.f,0.f,0.f,0.f,0.f,0.f,0.f};
  v8f acc[8];
#pragma unroll
  for (int i = 0; i < 8; ++i) acc[i] = vz;
  mm_tile(sA, sWT, lane, wv * 16, acc);

#pragma unroll
  for (int nt = 0; nt < 8; ++nt){
    const float bv_ = bias ? bias[nt * 16 + cb] : 0.f;
#pragma unroll
    for (int r = 0; r < 8; ++r){
      float x = acc[nt][r] + bv_;
      if (relu) x = fmaxf(x, 0.f);
      C[(m0 + wv * 16 + rofs + r) * 128 + nt * 16 + cb] = x;
    }
  }
}

// ---------------------------------------------------------------------------
// x0 = node_inputs @ W_init + b_init     (K=8, memory bound -> VALU)
// ---------------------------------------------------------------------------
__global__ __launch_bounds__(128) void init_embed(const float* __restrict__ ni,
                                                  const float* __restrict__ W,
                                                  const float* __restrict__ bias,
                                                  float* __restrict__ x0){
  __shared__ float row[8];
  const long n = blockIdx.x; const int c = threadIdx.x;
  if (c < 8) row[c] = ni[n * 8 + c];
  __syncthreads();
  float acc = bias[c];
#pragma unroll
  for (int j = 0; j < 8; ++j) acc += row[j] * W[j * 128 + c];
  x0[n * 128 + c] = acc;
}

__global__ void compute_deg(const int* __restrict__ adj, float* __restrict__ deg,
                            float* __restrict__ dinv){
  const int n = blockIdx.x * blockDim.x + threadIdx.x;
  if (n >= BNc) return;
  int cnt = 0;
#pragma unroll
  for (int k = 0; k < Kdeg; ++k) cnt += (adj[(long)n * Kdeg + k] >= 0);
  const float d = (float)cnt + 1.f;
  deg[n] = d; dinv[n] = rsqrtf(d);
}

// out[n] = sum_k enorm(n,k)*hw[adj] + hw[n]/deg[n] + b   (pure gather, K=16)
__global__ __launch_bounds__(128) void gcn_aggregate(const float* __restrict__ hw,
                                                     const int* __restrict__ adj,
                                                     const float* __restrict__ dinv,
                                                     const float* __restrict__ deg,
                                                     const float* __restrict__ bias,
                                                     float* __restrict__ out, int relu){
  __shared__ int   cols[Kdeg];
  __shared__ float wts[Kdeg];
  const long n = blockIdx.x; const int tid = threadIdx.x;
  const int b = (int)(n >> 10);
  if (tid < Kdeg){
    const int a = adj[n * Kdeg + tid];
    const int valid = a >= 0;
    const int col = b * Nnodes + (valid ? a : 0);
    cols[tid] = col;
    wts[tid]  = valid ? dinv[n] * dinv[col] : 0.f;
  }
  __syncthreads();
  float acc = hw[n * 128 + tid] / deg[n] + bias[tid];
#pragma unroll
  for (int k = 0; k < Kdeg; ++k) acc += wts[k] * hw[(long)cols[k] * 128 + tid];
  if (relu) acc = fmaxf(acc, 0.f);
  out[n * 128 + tid] = acc;
}

// y = maybe_relu( LN(alpha*a [+ beta*b]) * g + bt ), rows of 128
__global__ __launch_bounds__(128) void layernorm_k(const float* __restrict__ a,
                                                   const float* __restrict__ b,
                                                   float alpha, float beta,
                                                   const float* __restrict__ g,
                                                   const float* __restrict__ bt,
                                                   float* __restrict__ out, int relu){
  __shared__ float red[128];
  __shared__ float stat[2];
  const long r = blockIdx.x; const int c = threadIdx.x;
  float x = alpha * a[r * 128 + c];
  if (b) x += beta * b[r * 128 + c];
  red[c] = x; __syncthreads();
  for (int o = 64; o > 0; o >>= 1){ if (c < o) red[c] += red[c + o]; __syncthreads(); }
  if (c == 0) stat[0] = red[0] * (1.f / 128.f);
  __syncthreads();
  red[c] = x * x; __syncthreads();
  for (int o = 64; o > 0; o >>= 1){ if (c < o) red[c] += red[c + o]; __syncthreads(); }
  if (c == 0){ const float m = stat[0]; stat[1] = rsqrtf(red[0] * (1.f/128.f) - m*m + 1e-5f); }
  __syncthreads();
  float y = (x - stat[0]) * stat[1] * g[c] + bt[c];
  if (relu) y = fmaxf(y, 0.f);
  out[r * 128 + c] = y;
}

__global__ void zero_f32(float* p, int n){
  const int i = blockIdx.x * blockDim.x + threadIdx.x;
  if (i < n) p[i] = 0.f;
}

__global__ void mix_k(const float* __restrict__ gnn, const float* __restrict__ z,
                      float* __restrict__ out, int n){
  const int i = blockIdx.x * blockDim.x + threadIdx.x;
  if (i < n) out[i] = 0.8f * gnn[i] + 0.2f * z[i];
}

// ---------------------------------------------------------------------------
// SGFormer pass A: per 128-node tile, recompute K/V/Q heads with WMMA and
// accumulate  kvs[h] += K^T V  (second WMMA from transposed LDS tiles),
// ksum, ||q||^2, ||k||^2  via global float atomics.  Q/K/V never touch HBM.
// ---------------------------------------------------------------------------
__global__ __launch_bounds__(256) void sg_kv_accum(const float* __restrict__ z0,
    const float* __restrict__ Wq, const float* __restrict__ bq,
    const float* __restrict__ Wk, const float* __restrict__ bk,
    const float* __restrict__ Wv, const float* __restrict__ bv,
    float* __restrict__ kvs, float* __restrict__ ksum,
    float* __restrict__ ssq, float* __restrict__ ssk){
  extern __shared__ unsigned short sm[];
  unsigned short* z0t = sm;                // [node][k]   row-major
  unsigned short* wbT = sm + TILE_U;       // [col][k]    transposed weight panel
  unsigned short* ktT = sm + 2 * TILE_U;   // [m][node]   K^T, ready as A operand
  unsigned short* vtT = sm + 3 * TILE_U;   // [d][node]   V^T, ready as B operand
  const int tid = threadIdx.x, lane = tid & 31, wv = tid >> 5;
  const int cb = lane & 15, rofs = (lane >> 4) * 8;
  const long m0 = (long)blockIdx.x * 128;
  const v8f vz = {0.f,0.f,0.f,0.f,0.f,0.f,0.f,0.f};

  stage_rowmajor(z0t, z0 + m0 * 128, 128, tid);

  for (int h = 0; h < Hc; ++h){
    if (h + 1 < Hc){                        // hint next head's weight panels
      __builtin_prefetch(Wk + (h + 1) * 128, 0, 1);
      __builtin_prefetch(Wv + (h + 1) * 128, 0, 1);
      __builtin_prefetch(Wq + (h + 1) * 128, 0, 1);
    }
    // ---- K head tile -> ktT, ksum, ||k||^2 ----
    __syncthreads();
    stage_colmajor(wbT, Wk + h * 128, 512, tid);
    __syncthreads();
    v8f acc[8];
#pragma unroll
    for (int i = 0; i < 8; ++i) acc[i] = vz;
    mm_tile(z0t, wbT, lane, wv * 16, acc);
    float lss = 0.f;
#pragma unroll
    for (int nt = 0; nt < 8; ++nt){
      const float bia = bk[h * 128 + nt * 16 + cb];
      float csum = 0.f;
#pragma unroll
      for (int r = 0; r < 8; ++r){
        const float x = acc[nt][r] + bia;
        csum += x; lss += x * x;
        acc[nt][r] = x;
      }
      atomicAdd(&ksum[h * 128 + nt * 16 + cb], csum);
    }
    atomicAdd(ssk, lss);
    store_cT(ktT, acc, nullptr, wv, lane);
    // ---- V head tile -> vtT ----
    __syncthreads();
    stage_colmajor(wbT, Wv + h * 128, 512, tid);
    __syncthreads();
    v8f vac[8];
#pragma unroll
    for (int i = 0; i < 8; ++i) vac[i] = vz;
    mm_tile(z0t, wbT, lane, wv * 16, vac);
    store_cT(vtT, vac, bv + h * 128, wv, lane);
    // ---- Q head tile -> ||q||^2 only ----
    __syncthreads();
    stage_colmajor(wbT, Wq + h * 128, 512, tid);
    __syncthreads();
    v8f qac[8];
#pragma unroll
    for (int i = 0; i < 8; ++i) qac[i] = vz;
    mm_tile(z0t, wbT, lane, wv * 16, qac);
    float lsq = 0.f;
#pragma unroll
    for (int nt = 0; nt < 8; ++nt){
      const float bia = bq[h * 128 + nt * 16 + cb];
#pragma unroll
      for (int r = 0; r < 8; ++r){ const float x = qac[nt][r] + bia; lsq += x * x; }
    }
    atomicAdd(ssq, lsq);
    // ---- kvs[h] += K^T @ V :  A = ktT [m][node], B = vtT [d][node] ----
    __syncthreads();
    v8f kv[8];
#pragma unroll
    for (int i = 0; i < 8; ++i) kv[i] = vz;
    mm_tile(ktT, vtT, lane, wv * 16, kv);
#pragma unroll
    for (int nt = 0; nt < 8; ++nt)
#pragma unroll
      for (int r = 0; r < 8; ++r)
        atomicAdd(&kvs[(h * 128 + wv * 16 + rofs + r) * 128 + nt * 16 + cb], kv[nt][r]);
  }
}

// s = 1/(||q||*||k||); fold into bf16 kvs (stored TRANSPOSED [h][d][m]) + ksum.
__global__ void sg_finalize(const float* __restrict__ kvs, const float* __restrict__ ksum,
                            const float* __restrict__ ssq, const float* __restrict__ ssk,
                            unsigned short* __restrict__ kvs_bfT, float* __restrict__ ksum_s){
  const float s = rsqrtf(ssq[0] * ssk[0]);
  const int i = blockIdx.x * blockDim.x + threadIdx.x;
  if (i < Hc * Dc * Dc){
    const int h = i >> 14, m = (i >> 7) & 127, d = i & 127;
    kvs_bfT[(h << 14) + d * 128 + m] = f2bf(kvs[i] * s);
  }
  if (i < Hc * Dc) ksum_s[i] = ksum[i] * s;
}

// ---------------------------------------------------------------------------
// SGFormer pass B: per 128-node tile, recompute Q/V heads, then
// attn = mean_h[(q @ (s*kvs) + Nf*v) / (q . (s*ksum) + Nf)];  out = .5*attn+.5*z0
// ---------------------------------------------------------------------------
__global__ __launch_bounds__(256) void sg_attn_apply(const float* __restrict__ z0,
    const float* __restrict__ Wq, const float* __restrict__ bq,
    const float* __restrict__ Wv, const float* __restrict__ bv,
    const unsigned short* __restrict__ kvs_bfT, const float* __restrict__ ksum_s,
    float* __restrict__ out){
  extern __shared__ unsigned short sm[];
  unsigned short* z0t = sm;                // [node][k]  row-major
  unsigned short* wbT = sm + TILE_U;       // [col][k]   transposed weight panel
  unsigned short* qt  = sm + 2 * TILE_U;   // [node][m]  row-major (A of q@kvs)
  unsigned short* kvb = sm + 3 * TILE_U;   // [d][m]     transposed (B of q@kvs)
  float* den = (float*)(sm + 4 * TILE_U);
  const int tid = threadIdx.x, lane = tid & 31, wv = tid >> 5;
  const int cb = lane & 15, rofs = (lane >> 4) * 8;
  const long m0 = (long)blockIdx.x * 128;
  const float Nf = 65536.f;
  const v8f vz = {0.f,0.f,0.f,0.f,0.f,0.f,0.f,0.f};

  stage_rowmajor(z0t, z0 + m0 * 128, 128, tid);
  v8f att[8];
#pragma unroll
  for (int i = 0; i < 8; ++i) att[i] = vz;

  for (int h = 0; h < Hc; ++h){
    if (h + 1 < Hc){
      __builtin_prefetch(Wq + (h + 1) * 128, 0, 1);
      __builtin_prefetch(Wv + (h + 1) * 128, 0, 1);
      __builtin_prefetch(kvs_bfT + ((h + 1) << 14), 0, 1);
    }
    __syncthreads();
    stage_colmajor(wbT, Wq + h * 128, 512, tid);
    __syncthreads();
    v8f qa[8];
#pragma unroll
    for (int i = 0; i < 8; ++i) qa[i] = vz;
    mm_tile(z0t, wbT, lane, wv * 16, qa);
#pragma unroll
    for (int nt = 0; nt < 8; ++nt){
      const float bia = bq[h * 128 + nt * 16 + cb];
#pragma unroll
      for (int r = 0; r < 8; ++r)
        qt[(wv * 16 + rofs + r) * ST + nt * 16 + cb] = f2bf(qa[nt][r] + bia);
    }
    __syncthreads();                      // qt visible
    if (tid < 128){
      float s = Nf;
      for (int m = 0; m < 128; ++m) s += bf2f(qt[tid * ST + m]) * ksum_s[h * 128 + m];
      den[tid] = s;
    }
    stage_colmajor(wbT, Wv + h * 128, 512, tid);
    stage_u16(kvb, kvs_bfT + (h << 14), tid);
    __syncthreads();                      // den/wbT/kvb ready
    v8f va[8];
#pragma unroll
    for (int i = 0; i < 8; ++i) va[i] = vz;
    mm_tile(z0t, wbT, lane, wv * 16, va);
#pragma unroll
    for (int nt = 0; nt < 8; ++nt){
      const float bia = bv[h * 128 + nt * 16 + cb];
      v8f qk = vz;
#pragma unroll
      for (int kk = 0; kk < 128; kk += 32){
        v16bf a = load_a(qt, wv * 16, kk, lane);
        v16bf b = load_bT(kvb, kk, nt * 16, lane);
        qk = wmma_bf16(a, b, qk);
      }
#pragma unroll
      for (int r = 0; r < 8; ++r){
        const int row = wv * 16 + rofs + r;
        const float rd = 1.f / den[row];
        att[nt][r] += (qk[r] + Nf * (va[nt][r] + bia)) * rd;
      }
    }
  }
  __syncthreads();
#pragma unroll
  for (int nt = 0; nt < 8; ++nt)
#pragma unroll
    for (int r = 0; r < 8; ++r){
      const int row = wv * 16 + rofs + r, col = nt * 16 + cb;
      const float z0v = bf2f(z0t[row * ST + col]);
      out[(m0 + row) * 128 + col] = 0.125f * att[nt][r] + 0.5f * z0v;  // .5*mean4 + .5*z0
    }
}

// ---------------------------------------------------------------------------
// Decoder helpers (tiny M=64 problems -> VALU kernels)
// ---------------------------------------------------------------------------
__global__ __launch_bounds__(128) void gather_cur(const float* __restrict__ enh,
                                                  const int* __restrict__ curidx,
                                                  float* __restrict__ cur){
  const int b = blockIdx.x, c = threadIdx.x;
  const int idx = curidx[b];
  cur[b * 128 + c] = enh[((long)b * Nnodes + idx) * 128 + c];
}

__global__ void small_gemm(const float* __restrict__ A, const float* __restrict__ W,
                           const float* __restrict__ bias, float* __restrict__ C,
                           int Kd, int NcN, int relu){
  extern __shared__ float arow[];
  const int m = blockIdx.x;
  const int c = blockIdx.y * 128 + threadIdx.x;
  for (int j = threadIdx.x; j < Kd; j += 128) arow[j] = A[(long)m * Kd + j];
  __syncthreads();
  float acc = bias ? bias[c] : 0.f;
  for (int j = 0; j < Kd; ++j) acc += arow[j] * W[(long)j * NcN + c];
  if (relu) acc = fmaxf(acc, 0.f);
  C[(long)m * NcN + c] = acc;
}

__global__ __launch_bounds__(256) void dec_attn(const float* __restrict__ qd,
                                                const float* __restrict__ kd,
                                                const float* __restrict__ vd,
                                                float* __restrict__ att){
  __shared__ float sc[1024];
  __shared__ float qv[32];
  __shared__ float red[256];
  __shared__ float part[256];
  const int bh = blockIdx.x, b = bh >> 2, h = bh & 3, tid = threadIdx.x;
  if (tid < 32) qv[tid] = qd[b * 128 + h * 32 + tid];
  __syncthreads();
  const float scale = 0.17677669529663687f;  // 1/sqrt(32)
  for (int n = tid; n < 1024; n += 256){
    const float* kr = kd + ((long)(b * Nnodes + n)) * 128 + h * 32;
    float s = 0.f;
#pragma unroll
    for (int e = 0; e < 32; ++e) s += qv[e] * kr[e];
    sc[n] = s * scale;
  }
  __syncthreads();
  float m = -1e30f;
  for (int n = tid; n < 1024; n += 256) m = fmaxf(m, sc[n]);
  red[tid] = m; __syncthreads();
  for (int o = 128; o > 0; o >>= 1){ if (tid < o) red[tid] = fmaxf(red[tid], red[tid + o]); __syncthreads(); }
  const float mx = red[0]; __syncthreads();
  float ls = 0.f;
  for (int n = tid; n < 1024; n += 256){ const float e = __expf(sc[n] - mx); sc[n] = e; ls += e; }
  red[tid] = ls; __syncthreads();
  for (int o = 128; o > 0; o >>= 1){ if (tid < o) red[tid] += red[tid + o]; __syncthreads(); }
  const float denom = red[0]; __syncthreads();
  const int e = tid & 31, chunk = tid >> 5;
  float p = 0.f;
  for (int n = chunk * 128; n < chunk * 128 + 128; ++n)
    p += sc[n] * vd[((long)(b * Nnodes + n)) * 128 + h * 32 + e];
  part[tid] = p; __syncthreads();
  if (tid < 32){
    float s = 0.f;
#pragma unroll
    for (int ch = 0; ch < 8; ++ch) s += part[ch * 32 + tid];
    att[b * 128 + h * 32 + tid] = s / denom;
  }
}

__global__ __launch_bounds__(128) void qhead(const float* __restrict__ glob,
                                             const float* __restrict__ cur,
                                             const float* __restrict__ enh,
                                             const int* __restrict__ vps,
                                             const float* __restrict__ qW,
                                             const float* __restrict__ qb,
                                             float* __restrict__ out){
  __shared__ float gl[128], cu[128], w[384];
  const int b = blockIdx.x, t = threadIdx.x;
  gl[t] = glob[b * 128 + t]; cu[t] = cur[b * 128 + t];
  w[t] = qW[t]; w[128 + t] = qW[128 + t]; w[256 + t] = qW[256 + t];
  __syncthreads();
  const int idx = vps[b * VPc + t];
  const float* vr = enh + ((long)b * Nnodes + idx) * 128;
  float acc = qb[0];
  for (int j = 0; j < 128; ++j)
    acc += gl[j] * w[j] + cu[j] * w[128 + j] + vr[j] * w[256 + j];
  out[b * VPc + t] = acc;
}

// ---------------------------------------------------------------------------
extern "C" void kernel_launch(void* const* d_in, const int* in_sizes, int n_in,
                              void* d_out, int out_size, void* d_ws, size_t ws_size,
                              hipStream_t stream) {
  (void)in_sizes; (void)n_in; (void)out_size; (void)ws_size;
  const float* node_inputs = (const float*)d_in[0];
  const int*   cur_idx     = (const int*)  d_in[2];
  const int*   viewpoints  = (const int*)  d_in[3];
  const int*   adj         = (const int*)  d_in[5];
  const float* W_init = (const float*)d_in[6];
  const float* b_init = (const float*)d_in[7];
  const float* gcn_W  = (const float*)d_in[8];
  const float* gcn_b  = (const float*)d_in[9];
  const float* tc_fc_W = (const float*)d_in[10];
  const float* tc_fc_b = (const float*)d_in[11];
  const float* tc_ln1_g = (const float*)d_in[12];
  const float* tc_ln1_b = (const float*)d_in[13];
  const float* Wq = (const float*)d_in[14]; const float* bq = (const float*)d_in[15];
  const float* Wk = (const float*)d_in[16]; const float* bk = (const float*)d_in[17];
  const float* Wv = (const float*)d_in[18]; const float* bv = (const float*)d_in[19];
  const float* tc_ln2_g = (const float*)d_in[20];
  const float* tc_ln2_b = (const float*)d_in[21];
  const float* sg_fc_W = (const float*)d_in[22];
  const float* sg_fc_b = (const float*)d_in[23];
  const float* dec_Wq = (const float*)d_in[24];
  const float* dec_Wk = (const float*)d_in[25];
  const float* dec_Wv = (const float*)d_in[26];
  const float* dec_Wo = (const float*)d_in[27];
  const float* dec_ln1_g = (const float*)d_in[28];
  const float* dec_ln1_b = (const float*)d_in[29];
  const float* ff_W1 = (const float*)d_in[30]; const float* ff_b1 = (const float*)d_in[31];
  const float* ff_W2 = (const float*)d_in[32]; const float* ff_b2 = (const float*)d_in[33];
  const float* dec_ln2_g = (const float*)d_in[34];
  const float* dec_ln2_b = (const float*)d_in[35];
  const float* q_W = (const float*)d_in[36]; const float* q_b = (const float*)d_in[37];

  // ---- workspace carving (floats) ----
  float* ws = (float*)d_ws;
  const size_t T = (size_t)BNc * Dc;           // 8,388,608 floats per big buffer
  float* W0 = ws;                              // x0 -> zpre/z -> vd
  float* W1 = ws + T;                          // GCN ping -> kd
  float* W2 = ws + 2 * T;                      // GCN pong / gnn_out -> enhanced
  float* W3 = ws + 3 * T;                      // GEMM scratch / mix
  float* W4 = ws + 4 * T;                      // z0
  float* p  = ws + 5 * T;
  float* deg  = p; p += BNc;
  float* dinv = p; p += BNc;
  float* kvs  = p; p += Hc * Dc * Dc;          // 65536
  float* ksum = p; p += Hc * Dc;               // 512
  float* ksum_s = p; p += Hc * Dc;             // 512
  float* ssq  = p; p += 1;
  float* ssk  = p; p += 1;
  p += 2;                                      // keep 16B alignment
  unsigned short* kvs_bfT = (unsigned short*)p; p += (Hc * Dc * Dc) / 2;
  float* cur    = p; p += Bc * Dc;
  float* qd     = p; p += Bc * Dc;
  float* attb   = p; p += Bc * Dc;
  float* attprj = p; p += Bc * Dc;
  float* tbuf   = p; p += Bc * Dc;
  float* ffbuf  = p; p += Bc * FFc;
  float* ffout  = p; p += Bc * Dc;
  float* glob   = p; p += Bc * Dc;

  const size_t lds2 = 2 * TILE_U * sizeof(unsigned short);            // gemm128
  const size_t lds4 = 4 * TILE_U * sizeof(unsigned short);            // kv_accum
  const size_t lds4d = lds4 + 128 * sizeof(float) + 16;               // attn_apply

  // ---- initial embedding + degrees ----
  init_embed<<<BNc, 128, 0, stream>>>(node_inputs, W_init, b_init, W0);
  compute_deg<<<BNc / 256, 256, 0, stream>>>(adj, deg, dinv);

  // ---- GCN: 4 x (WMMA gemm + gather aggregate) ----
  const float* hin = W0; float* hout = W1;
  for (int i = 0; i < 4; ++i){
    gemm128_bf16<<<BNc / 128, 256, lds2, stream>>>(hin, gcn_W + (size_t)i * Dc * Dc,
                                                   nullptr, W3, 0);
    gcn_aggregate<<<BNc, 128, 0, stream>>>(W3, adj, dinv, deg, gcn_b + i * Dc, hout,
                                           i < 3 ? 1 : 0);
    hin = hout; hout = (hout == W1) ? W2 : W1;
  }
  // gnn_out lands in W2.

  // ---- SGFormer TransConv ----
  gemm128_bf16<<<BNc / 128, 256, lds2, stream>>>(W0, tc_fc_W, tc_fc_b, W3, 0);
  layernorm_k<<<BNc, 128, 0, stream>>>(W3, nullptr, 1.f, 0.f, tc_ln1_g, tc_ln1_b, W4, 1);

  zero_f32<<<(Hc*Dc*Dc + 2*Hc*Dc + 4 + 255) / 256, 256, 0, stream>>>(kvs,
            Hc*Dc*Dc + 2*Hc*Dc + 4);
  sg_kv_accum<<<BNc / 128, 256, lds4, stream>>>(W4, Wq, bq, Wk, bk, Wv, bv,
                                                kvs, ksum, ssq, ssk);
  sg_finalize<<<(Hc*Dc*Dc + 255) / 256, 256, 0, stream>>>(kvs, ksum, ssq, ssk,
                                                          kvs_bfT, ksum_s);
  sg_attn_apply<<<BNc / 128, 256, lds4d, stream>>>(W4, Wq, bq, Wv, bv,
                                                   kvs_bfT, ksum_s, W0);
  layernorm_k<<<BNc, 128, 0, stream>>>(W0, nullptr, 1.f, 0.f, tc_ln2_g, tc_ln2_b, W0, 1);

  // ---- enhanced = (0.8*gnn + 0.2*z) @ sg_fc_W + b ----
  mix_k<<<(int)(T / 256), 256, 0, stream>>>(W2, W0, W3, (int)T);
  gemm128_bf16<<<BNc / 128, 256, lds2, stream>>>(W3, sg_fc_W, sg_fc_b, W2, 0);

  // ---- decoder ----
  gather_cur<<<Bc, 128, 0, stream>>>(W2, cur_idx, cur);
  gemm128_bf16<<<BNc / 128, 256, lds2, stream>>>(W2, dec_Wk, nullptr, W1, 0);   // kd
  gemm128_bf16<<<BNc / 128, 256, lds2, stream>>>(W2, dec_Wv, nullptr, W0, 0);   // vd
  small_gemm<<<dim3(Bc, 1), 128, 128 * sizeof(float), stream>>>(cur, dec_Wq, nullptr,
                                                                qd, 128, 128, 0);
  dec_attn<<<Bc * 4, 256, 0, stream>>>(qd, W1, W0, attb);
  small_gemm<<<dim3(Bc, 1), 128, 128 * sizeof(float), stream>>>(attb, dec_Wo, nullptr,
                                                                attprj, 128, 128, 0);
  layernorm_k<<<Bc, 128, 0, stream>>>(attprj, cur, 1.f, 1.f, dec_ln1_g, dec_ln1_b, tbuf, 0);
  small_gemm<<<dim3(Bc, 4), 128, 128 * sizeof(float), stream>>>(tbuf, ff_W1, ff_b1,
                                                                ffbuf, 128, 512, 1);
  small_gemm<<<dim3(Bc, 1), 128, 512 * sizeof(float), stream>>>(ffbuf, ff_W2, ff_b2,
                                                                ffout, 512, 128, 0);
  layernorm_k<<<Bc, 128, 0, stream>>>(ffout, tbuf, 1.f, 1.f, dec_ln2_g, dec_ln2_b, glob, 0);

  // ---- Q head ----
  qhead<<<Bc, 128, 0, stream>>>(glob, cur, W2, viewpoints, q_W, q_b, (float*)d_out);
}
